// SpatialProcessor_66116726555145
// MI455X (gfx1250) — matrix operand order repaired and checked by reading, not compiled, not run on hardware
//
#include <hip/hip_runtime.h>
#include <cstdint>

typedef __attribute__((ext_vector_type(16))) _Float16 v16h;
typedef __attribute__((ext_vector_type(8)))  float    v8f;

#define NNODES 4096
#define EMB    16
#define FDIM   3
#define NH1    4
#define CDIM   128
#define DCAT   512      // NH1*CDIM
#define MW     128      // mask words per row (4096/32)
#define JTILE  32
#define NJT    128      // 4096/32
#define NCH    8        // 128/16 column chunks
#define NIT    256      // 4096/16 i-tiles
#define JS1    2
#define JS2    8

// ---------- block reductions (blockDim.x == 128) ----------
__device__ __forceinline__ float blockReduceSum128(float v, float* sh) {
  int t = threadIdx.x;
  sh[t] = v; __syncthreads();
  #pragma unroll
  for (int s = 64; s > 0; s >>= 1) { if (t < s) sh[t] += sh[t + s]; __syncthreads(); }
  float r = sh[0]; __syncthreads();
  return r;
}
__device__ __forceinline__ float blockReduceMax128(float v, float* sh) {
  int t = threadIdx.x;
  sh[t] = v; __syncthreads();
  #pragma unroll
  for (int s = 64; s > 0; s >>= 1) { if (t < s) sh[t] = fmaxf(sh[t], sh[t + s]); __syncthreads(); }
  float r = sh[0]; __syncthreads();
  return r;
}

// ---------- 1) adjacency bitmask: mask(i,j) = (i==j) || dot(emb_i,emb_j) > 0 ----------
__global__ void mask_kernel(const float* __restrict__ emb, uint32_t* __restrict__ maskw) {
  int i = blockIdx.x;
  int wave = threadIdx.x >> 5;
  int lane = threadIdx.x & 31;
  const float4* ei = (const float4*)(emb + (size_t)i * EMB);
  float4 e0 = ei[0], e1 = ei[1], e2 = ei[2], e3 = ei[3];
  for (int it = 0; it < 32; ++it) {
    int j = it * 128 + wave * 32 + lane;
    const float4* ej = (const float4*)(emb + (size_t)j * EMB);
    float4 a = ej[0], b = ej[1], c = ej[2], d = ej[3];
    float dot = e0.x*a.x + e0.y*a.y + e0.z*a.z + e0.w*a.w
              + e1.x*b.x + e1.y*b.y + e1.z*b.z + e1.w*b.w
              + e2.x*c.x + e2.y*c.y + e2.z*c.z + e2.w*c.w
              + e3.x*d.x + e3.y*d.y + e3.z*d.z + e3.w*d.w;
    bool on = (dot > 0.0f) || (j == i);
#if defined(__AMDGCN__) && __has_builtin(__builtin_amdgcn_ballot_w32)
    uint32_t word = __builtin_amdgcn_ballot_w32(on);      // wave32 ballot -> 32-bit mask word
#else
    uint32_t word = (uint32_t)__ballot(on);
#endif
    if (lane == 0) maskw[(size_t)i * MW + it * 4 + wave] = word;
  }
}

// ---------- 2) layer-1 prep: h1[h][n][c] = x@W1, src/dst logit halves ----------
__global__ void prep1_kernel(const float* __restrict__ x, const float* __restrict__ W1,
                             const float* __restrict__ a_self, const float* __restrict__ a_neigh,
                             float* __restrict__ h1, float* __restrict__ src, float* __restrict__ dst) {
  __shared__ float sh[128];
  int n = blockIdx.x, c = threadIdx.x;
  float x0 = x[n*FDIM+0], x1 = x[n*FDIM+1], x2 = x[n*FDIM+2];
  float hv[NH1];
  #pragma unroll
  for (int h = 0; h < NH1; ++h) {
    int wi = h * CDIM + c;                       // W1[f][h][c]
    float v = x0*W1[wi] + x1*W1[DCAT + wi] + x2*W1[2*DCAT + wi];
    hv[h] = v;
    h1[((size_t)h * NNODES + n) * CDIM + c] = v;
  }
  #pragma unroll
  for (int h = 0; h < NH1; ++h) {
    float s = blockReduceSum128(hv[h] * a_self[h*CDIM + c],  sh);
    float d = blockReduceSum128(hv[h] * a_neigh[h*CDIM + c], sh);
    if (c == 0) { src[h*NNODES + n] = s; dst[h*NNODES + n] = d; }
  }
}

// ---------- 3) pre-pack h into WMMA B-fragment layout (f16, 32x16 tiles) ----------
// B order used here (matches A): lane l -> column N=l&15; per-lane K slots = (l>>4)*16 + s, s=0..15
// (packed as f16 pairs: uint32 = {K odd | K even}).  Storage: two uint4 per lane per chunk.
__global__ void pack_kernel(const float* __restrict__ h, uint4* __restrict__ pack, int H) {
  int t = blockIdx.x * 256 + threadIdx.x;       // H * 65536 threads
  int w    =  t        & 1;
  int lane = (t >> 1)  & 31;
  int ch   = (t >> 6)  & 7;
  int jt   = (t >> 9)  & 127;
  int head =  t >> 16;
  int half = lane >> 4, nn = lane & 15;
  int c  = ch * 16 + nn;
  int kb = half * 16 + w * 8;
  int j0 = jt * JTILE;
  uint32_t vv[4];
  #pragma unroll
  for (int q = 0; q < 4; ++q) {
    float f0 = h[((size_t)head*NNODES + j0 + kb + 2*q    ) * CDIM + c];
    float f1 = h[((size_t)head*NNODES + j0 + kb + 2*q + 1) * CDIM + c];
    union { _Float16 f; uint16_t u; } u0, u1;
    u0.f = (_Float16)f0; u1.f = (_Float16)f1;
    vv[q] = (uint32_t)u0.u | ((uint32_t)u1.u << 16);
  }
  pack[((size_t)(head*NJT + jt)*NCH + ch)*64 + lane*2 + w] = make_uint4(vv[0], vv[1], vv[2], vv[3]);
}

// ---------- 4) exact softmax row stats: M = row max, D = sum exp(e-M) over masked j ----------
__global__ void stats_kernel(const float* __restrict__ src, const float* __restrict__ dst,
                             const uint32_t* __restrict__ maskw,
                             float* __restrict__ Mv, float* __restrict__ Dv) {
  __shared__ float sh[128];
  int head = blockIdx.x >> 12;
  int i = blockIdx.x & (NNODES - 1);
  float si = src[(size_t)head*NNODES + i];
  const float* drow = dst + (size_t)head*NNODES;
  const uint32_t* mrow = maskw + (size_t)i * MW;
  int t = threadIdx.x;
  float mx = -1e30f;
  for (int j = t; j < NNODES; j += 128) {
    float bit = (float)((mrow[j >> 5] >> (j & 31)) & 1u);
    float s = si + drow[j]; s = s > 0.f ? s : 0.2f * s;
    mx = fmaxf(mx, bit != 0.f ? s : -1e30f);
  }
  mx = blockReduceMax128(mx, sh);
  float sum = 0.f;
  for (int j = t; j < NNODES; j += 128) {
    float bit = (float)((mrow[j >> 5] >> (j & 31)) & 1u);
    float s = si + drow[j]; s = s > 0.f ? s : 0.2f * s;
    // masked-in entries have s-mx <= 0, so the clamp is exact; it only guards
    // masked-out lanes against exp overflow before the *0 kill.
    sum += __expf(fminf(s - mx, 0.f)) * bit;
  }
  sum = blockReduceSum128(sum, sh);
  if (t == 0) { Mv[(size_t)head*NNODES + i] = mx; Dv[(size_t)head*NNODES + i] = sum; }
}

// ---------- 5) fused attention-aggregate (double-buffered WMMA pipeline) ----------
// A order used here (matches B pack): lane l -> row M=l&15; slots t: K=(l>>4)*16+t
// Accumulates unnormalized sum( exp(e-M) * h ) with v_wmma_f32_16x16x32_f16; epilogue divides by D.

__device__ __forceinline__ void loadB16(const uint4* __restrict__ pt, uint4 (&buf)[16]) {
  #pragma unroll
  for (int ch = 0; ch < NCH; ++ch) {
    buf[2*ch]     = pt[ch * 64];
    buf[2*ch + 1] = pt[ch * 64 + 1];
  }
}

__device__ __forceinline__ void wmma8(const v16h& a, const uint4 (&buf)[16], v8f (&acc)[NCH]) {
  #pragma unroll
  for (int ch = 0; ch < NCH; ++ch) {
    union { uint4 u[2]; v16h v; } bb;
    bb.u[0] = buf[2*ch];
    bb.u[1] = buf[2*ch + 1];
    acc[ch] = __builtin_amdgcn_wmma_f32_16x16x32_f16(
        false, a, false, bb.v, (short)0, acc[ch], false, false);
  }
}

__global__ __launch_bounds__(32) void attn_kernel(
    const float* __restrict__ src, const float* __restrict__ dst, const float* __restrict__ Mv,
    const uint32_t* __restrict__ maskw, const uint4* __restrict__ pack,
    float* __restrict__ part, int H, int JS) {
  int itile = blockIdx.x, head = blockIdx.y, js = blockIdx.z;
  int lane = threadIdx.x;
  int half = lane >> 4, mrow = lane & 15;
  int i = itile * 16 + mrow;
  float si = src[(size_t)head*NNODES + i];
  float Mi = Mv[(size_t)head*NNODES + i];
  const float* drow = dst + (size_t)head*NNODES;
  const uint32_t* mwrow = maskw + (size_t)i * MW;

  // builds the P (=exp(e-M), masked) fragment for one 16x32 j-tile in A layout.
  // Fully branchless: kept entries satisfy s-Mi <= 0 so the clamp is exact;
  // the clamp only protects dropped entries from exp overflow before the *0.
  auto buildA = [&](int jt) -> v16h {
    int j0 = jt * JTILE;
    float dstv = drow[j0 + lane];                 // 32 neighbor logit halves across lanes
    uint32_t mw = mwrow[jt];                      // 32 mask bits for this lane's row
    v16h a;
    #pragma unroll
    for (int t = 0; t < 16; ++t) {
      int k = half * 16 + t;
      float dk = __shfl(dstv, k, 32);
      float s = si + dk;
      s = s > 0.f ? s : 0.2f * s;                 // LeakyReLU(0.2)
      float p = __expf(fminf(s - Mi, 0.f)) * (float)((mw >> k) & 1u);
      a[t] = (_Float16)p;
    }
    return a;
  };

  v8f acc[NCH];
  #pragma unroll
  for (int ch = 0; ch < NCH; ++ch) acc[ch] = {};

  const uint4* base = pack + (size_t)head * NJT * 512 + lane * 2;
  int jt_per = NJT / JS;                          // 64 (layer1) or 16 (layer2): always even
  int jt0 = js * jt_per, jt_end = jt0 + jt_per;

  // software pipeline: while WMMAs for tile t run, B loads + A build for tile t+1 are in flight
  uint4 buf0[16], buf1[16];
  v16h a0, a1;
  loadB16(base + (size_t)jt0 * 512, buf0);
  a0 = buildA(jt0);
  for (int jt = jt0; jt < jt_end; jt += 2) {
    loadB16(base + (size_t)(jt + 1) * 512, buf1);
    a1 = buildA(jt + 1);
    wmma8(a0, buf0, acc);
    if (jt + 2 < jt_end) {
      loadB16(base + (size_t)(jt + 2) * 512, buf0);
      a0 = buildA(jt + 2);
    }
    wmma8(a1, buf1, acc);
  }

  // C/D layout: VGPR v, row = v + 8*(lane>>4), col = lane&15 within chunk
  float* pout = part + (size_t)(js * H + head) * NNODES * CDIM;
  #pragma unroll
  for (int ch = 0; ch < NCH; ++ch)
    #pragma unroll
    for (int v = 0; v < 8; ++v) {
      int row = itile * 16 + v + 8 * half;
      pout[(size_t)row * CDIM + ch * 16 + mrow] = acc[ch][v];
    }
}

// ---------- 6) layer-1 epilogue: sum partials, /D, +bias, ReLU, concat heads ----------
__global__ void reduce1_kernel(const float* __restrict__ part, const float* __restrict__ Dv,
                               const float* __restrict__ b1, float* __restrict__ h1cat) {
  int n = blockIdx.x, t = threadIdx.x;                  // 512 threads: t = head*128 + c
  int head = t >> 7, c = t & 127;
  float s = 0.f;
  #pragma unroll
  for (int js = 0; js < JS1; ++js)
    s += part[((size_t)(js * NH1 + head) * NNODES + n) * CDIM + c];
  float v = s / Dv[(size_t)head*NNODES + n] + b1[t];
  h1cat[(size_t)n * DCAT + t] = v > 0.f ? v : 0.f;
}

// ---------- 7) layer-2 prep: h2 = h1cat @ W2, src2/dst2 ----------
__global__ void prep2_kernel(const float* __restrict__ h1cat, const float* __restrict__ W2,
                             const float* __restrict__ a_self, const float* __restrict__ a_neigh,
                             float* __restrict__ h2, float* __restrict__ src, float* __restrict__ dst) {
  __shared__ float row[DCAT];
  __shared__ float sh[128];
  int n = blockIdx.x, c = threadIdx.x;
  #pragma unroll
  for (int q = 0; q < 4; ++q) row[c + q * 128] = h1cat[(size_t)n * DCAT + c + q * 128];
  __syncthreads();
  float acc = 0.f;
  for (int k = 0; k < DCAT; ++k) acc += row[k] * W2[k * CDIM + c];
  h2[(size_t)n * CDIM + c] = acc;
  float s = blockReduceSum128(acc * a_self[c],  sh);
  float d = blockReduceSum128(acc * a_neigh[c], sh);
  if (c == 0) { src[n] = s; dst[n] = d; }
}

// ---------- 8) layer-2 epilogue ----------
__global__ void reduce2_kernel(const float* __restrict__ part, const float* __restrict__ Dv,
                               const float* __restrict__ b2, float* __restrict__ out) {
  int n = blockIdx.x, c = threadIdx.x;
  float s = 0.f;
  #pragma unroll
  for (int js = 0; js < JS2; ++js)
    s += part[((size_t)js * NNODES + n) * CDIM + c];
  out[(size_t)n * CDIM + c] = s / Dv[n] + b2[c];        // mean over H=1 heads == identity
}

extern "C" void kernel_launch(void* const* d_in, const int* in_sizes, int n_in,
                              void* d_out, int out_size, void* d_ws, size_t ws_size,
                              hipStream_t stream) {
  (void)in_sizes; (void)n_in; (void)out_size; (void)ws_size;
  const float* x        = (const float*)d_in[0];
  const float* emb      = (const float*)d_in[1];
  const float* W1       = (const float*)d_in[2];
  const float* a_self1  = (const float*)d_in[3];
  const float* a_neigh1 = (const float*)d_in[4];
  const float* b1       = (const float*)d_in[5];
  const float* W2       = (const float*)d_in[6];
  const float* a_self2  = (const float*)d_in[7];
  const float* a_neigh2 = (const float*)d_in[8];
  const float* b2       = (const float*)d_in[9];
  float* out = (float*)d_out;

  // workspace carve-out (~33.4 MB total; everything stays L2-resident on MI455X)
  char* w = (char*)d_ws;
  auto take = [&](size_t bytes) { char* p = w; w += (bytes + 255) & ~(size_t)255; return p; };
  uint32_t* maskw = (uint32_t*)take((size_t)NNODES * MW * 4);            // 2 MB
  float*    h1    = (float*)   take((size_t)NH1 * NNODES * CDIM * 4);    // 8 MB (reused as h1cat)
  uint4*    pack1 = (uint4*)   take((size_t)NH1 * NJT * NCH * 64 * 16);  // 4 MB
  float*    src1  = (float*)   take((size_t)NH1 * NNODES * 4);
  float*    dst1  = (float*)   take((size_t)NH1 * NNODES * 4);
  float*    M1    = (float*)   take((size_t)NH1 * NNODES * 4);
  float*    Dv1   = (float*)   take((size_t)NH1 * NNODES * 4);
  float*    part  = (float*)   take((size_t)8 * NNODES * CDIM * 4);      // 16 MB, both layers
  float*    h2    = (float*)   take((size_t)NNODES * CDIM * 4);          // 2 MB
  uint4*    pack2 = (uint4*)   take((size_t)NJT * NCH * 64 * 16);        // 1 MB
  float*    src2  = (float*)   take((size_t)NNODES * 4);
  float*    dst2  = (float*)   take((size_t)NNODES * 4);
  float*    M2    = (float*)   take((size_t)NNODES * 4);
  float*    Dv2   = (float*)   take((size_t)NNODES * 4);
  float*    h1cat = h1;  // safe alias: raw h1 is dead after pack1/stats1

  mask_kernel  <<<NNODES, 128, 0, stream>>>(emb, maskw);
  prep1_kernel <<<NNODES, 128, 0, stream>>>(x, W1, a_self1, a_neigh1, h1, src1, dst1);
  pack_kernel  <<<NH1 * 256, 256, 0, stream>>>(h1, pack1, NH1);
  stats_kernel <<<NH1 * NNODES, 128, 0, stream>>>(src1, dst1, maskw, M1, Dv1);
  attn_kernel  <<<dim3(NIT, NH1, JS1), 32, 0, stream>>>(src1, dst1, M1, maskw, pack1, part, NH1, JS1);
  reduce1_kernel<<<NNODES, 512, 0, stream>>>(part, Dv1, b1, h1cat);
  prep2_kernel <<<NNODES, 128, 0, stream>>>(h1cat, W2, a_self2, a_neigh2, h2, src2, dst2);
  pack_kernel  <<<256, 256, 0, stream>>>(h2, pack2, 1);
  stats_kernel <<<NNODES, 128, 0, stream>>>(src2, dst2, maskw, M2, Dv2);
  attn_kernel  <<<dim3(NIT, 1, JS2), 32, 0, stream>>>(src2, dst2, M2, maskw, pack2, part, 1, JS2);
  reduce2_kernel<<<NNODES, 128, 0, stream>>>(part, Dv2, b2, out);
}
